// UniversalAMRPALayer_57647051046970
// MI455X (gfx1250) — compile-verified, matrix operand order checked by reading, not run on hardware
//
#include <hip/hip_runtime.h>

#define SQ   8192
#define EMB  768
#define E3T  2304
#define DKC  64
#define NCH  26              // 832/32 augmented-head chunks
#define SCALE_ 0.125f
#define LOG2E_ 1.4426950408889634f

typedef __attribute__((ext_vector_type(16))) __bf16         v16bf;
typedef __attribute__((ext_vector_type(8)))  float          v8f;
typedef __attribute__((ext_vector_type(8)))  unsigned short u16x8;
typedef __attribute__((ext_vector_type(16))) unsigned short u16x16;

#define WMMA_BF16(A,B,C) \
  __builtin_amdgcn_wmma_f32_16x16x32_bf16(false,(A),false,(B),(short)0,(C),false,false)

__device__ __forceinline__ unsigned short f2bf(float f) {
  unsigned int u = __float_as_uint(f);
  unsigned int r = u + 0x7FFFu + ((u >> 16) & 1u);   // round-to-nearest-even
  return (unsigned short)(r >> 16);
}

// Load one 16x32 bf16 A/B fragment (row-major rows, ISA 16-bit operand layout):
// lane (lo,hi): row = row0+lo ; elems[0..7] = K hi*8.. ; elems[8..15] = K 16+hi*8..
__device__ __forceinline__ v16bf ld_frag(const unsigned short* base, int row0,
                                         int pitch, int k0, int lo, int hi) {
  const unsigned short* p = base + (size_t)(row0 + lo) * pitch + k0 + hi * 8;
  u16x8 x = *(const u16x8*)p;
  u16x8 y = *(const u16x8*)(p + 16);
  u16x16 v = __builtin_shufflevector(x, y, 0,1,2,3,4,5,6,7,8,9,10,11,12,13,14,15);
  return __builtin_bit_cast(v16bf, v);
}

__device__ __forceinline__ float hmax16(float v) {
  #pragma unroll
  for (int m = 1; m < 16; m <<= 1) v = fmaxf(v, __shfl_xor(v, m, 32));
  return v;
}
__device__ __forceinline__ float hsum16(float v) {
  #pragma unroll
  for (int m = 1; m < 16; m <<= 1) v += __shfl_xor(v, m, 32);
  return v;
}

// ---------------- conversion / transpose pre-passes ----------------
__global__ void k_cvt(const float* __restrict__ src, unsigned short* __restrict__ dst, int n) {
  int i = blockIdx.x * blockDim.x + threadIdx.x;
  if (i < n) dst[i] = f2bf(src[i]);
}
__global__ void k_cvt_tr(const float* __restrict__ src, unsigned short* __restrict__ dst,
                         int rows, int cols) {
  int i = blockIdx.x * blockDim.x + threadIdx.x;
  if (i < rows * cols) {
    int r = i / cols, c = i % cols;
    dst[(size_t)c * rows + r] = f2bf(src[i]);
  }
}

// ---------------- QKV projection: qkv = hs @ Wc + bc ----------------
// Routes cols 0..767 -> Qb (row-major), 768..1535 -> Kb (row-major),
// 1536..2303 -> Vt (transposed, dim-major [768][8192]).
__global__ void __launch_bounds__(128) k_qkv(const unsigned short* __restrict__ hsb,
                                             const unsigned short* __restrict__ Wct,
                                             const float* __restrict__ bc,
                                             unsigned short* __restrict__ Qb,
                                             unsigned short* __restrict__ Kb,
                                             unsigned short* __restrict__ Vt) {
  int lane = threadIdx.x & 31, wave = threadIdx.x >> 5;
  int gid = blockIdx.x * 4 + wave;
  int rt = gid / (E3T / 16), ct = gid % (E3T / 16);
  int row0 = rt * 16, col0 = ct * 16;
  int lo = lane & 15, hi = lane >> 4;
  v8f acc = {};
  #pragma unroll
  for (int c = 0; c < EMB / 32; ++c) {
    v16bf a = ld_frag(hsb, row0, EMB, c * 32, lo, hi);
    v16bf b = ld_frag(Wct, col0, EMB, c * 32, lo, hi);
    acc = WMMA_BF16(a, b, acc);
  }
  int col = col0 + lo;
  float bias = bc[col];
  #pragma unroll
  for (int j = 0; j < 8; ++j) {
    int row = row0 + j + 8 * hi;
    float v = acc[j] + bias;
    if (col < EMB)            Qb[(size_t)row * EMB + col]               = f2bf(v);
    else if (col < 2 * EMB)   Kb[(size_t)row * EMB + (col - EMB)]       = f2bf(v);
    else                      Vt[(size_t)(col - 2 * EMB) * SQ + row]    = f2bf(v);
  }
}

// ---------------- base attention (dim 64, NO causal mask, full S) ----------------
__global__ void __launch_bounds__(128) k_base_attn(const unsigned short* __restrict__ Qb,
                                                   const unsigned short* __restrict__ Kb,
                                                   const unsigned short* __restrict__ Vt,
                                                   unsigned short* __restrict__ bob) {
  __shared__ __align__(16) unsigned short pbuf[4][16 * 32];
  int lane = threadIdx.x & 31, wave = threadIdx.x >> 5;
  int q0 = (blockIdx.x * 4 + wave) * 16;
  int lo = lane & 15, hi = lane >> 4;
  v16bf aq0 = ld_frag(Qb, q0, EMB, 0,  lo, hi);
  v16bf aq1 = ld_frag(Qb, q0, EMB, 32, lo, hi);
  v8f o0 = {}, o1 = {}, o2 = {}, o3 = {};
  float m[8], l[8];
  #pragma unroll
  for (int j = 0; j < 8; ++j) { m[j] = -1e30f; l[j] = 0.f; }
  for (int kb = 0; kb < SQ; kb += 32) {
    v16bf b00 = ld_frag(Kb, kb,      EMB, 0,  lo, hi);
    v16bf b10 = ld_frag(Kb, kb + 16, EMB, 0,  lo, hi);
    v16bf b01 = ld_frag(Kb, kb,      EMB, 32, lo, hi);
    v16bf b11 = ld_frag(Kb, kb + 16, EMB, 32, lo, hi);
    v8f s0 = {}, s1 = {};
    s0 = WMMA_BF16(aq0, b00, s0); s0 = WMMA_BF16(aq1, b01, s0);
    s1 = WMMA_BF16(aq0, b10, s1); s1 = WMMA_BF16(aq1, b11, s1);
    float alpha[8];
    #pragma unroll
    for (int j = 0; j < 8; ++j) {
      float v0 = s0[j] * SCALE_, v1 = s1[j] * SCALE_;
      float mn = fmaxf(m[j], hmax16(fmaxf(v0, v1)));
      alpha[j] = exp2f((m[j] - mn) * LOG2E_);
      m[j] = mn;
      float p0 = exp2f((v0 - mn) * LOG2E_);
      float p1 = exp2f((v1 - mn) * LOG2E_);
      pbuf[wave][(j + 8 * hi) * 32 + lo]      = f2bf(p0);
      pbuf[wave][(j + 8 * hi) * 32 + 16 + lo] = f2bf(p1);
      l[j] = l[j] * alpha[j] + hsum16(p0 + p1);
    }
    asm volatile("s_wait_dscnt 0" ::: "memory");
    v16bf ap = ld_frag(&pbuf[wave][0], 0, 32, 0, lo, hi);
    #pragma unroll
    for (int j = 0; j < 8; ++j) { o0[j] *= alpha[j]; o1[j] *= alpha[j]; o2[j] *= alpha[j]; o3[j] *= alpha[j]; }
    v16bf bv0 = ld_frag(Vt, 0,  SQ, kb, lo, hi);
    v16bf bv1 = ld_frag(Vt, 16, SQ, kb, lo, hi);
    v16bf bv2 = ld_frag(Vt, 32, SQ, kb, lo, hi);
    v16bf bv3 = ld_frag(Vt, 48, SQ, kb, lo, hi);
    o0 = WMMA_BF16(ap, bv0, o0); o1 = WMMA_BF16(ap, bv1, o1);
    o2 = WMMA_BF16(ap, bv2, o2); o3 = WMMA_BF16(ap, bv3, o3);
  }
  #pragma unroll
  for (int j = 0; j < 8; ++j) {
    size_t row = q0 + j + 8 * hi;
    float inv = 1.f / l[j];
    bob[row * DKC + 0  + lo] = f2bf(o0[j] * inv);
    bob[row * DKC + 16 + lo] = f2bf(o1[j] * inv);
    bob[row * DKC + 32 + lo] = f2bf(o2[j] * inv);
    bob[row * DKC + 48 + lo] = f2bf(o3[j] * inv);
  }
}

// ---------------- mem = (base_A @ Vc) @ W_mem * LAYER_DEPTH ----------------
__global__ void __launch_bounds__(128) k_memproj(const unsigned short* __restrict__ bob,
                                                 const unsigned short* __restrict__ WmT,
                                                 unsigned short* __restrict__ memb) {
  int lane = threadIdx.x & 31, wave = threadIdx.x >> 5;
  int gid = blockIdx.x * 4 + wave;
  int rt = gid >> 2, ct = gid & 3;
  int row0 = rt * 16, col0 = ct * 16, lo = lane & 15, hi = lane >> 4;
  v8f acc = {};
  #pragma unroll
  for (int c = 0; c < 2; ++c) {
    v16bf a = ld_frag(bob, row0, DKC, c * 32, lo, hi);
    v16bf b = ld_frag(WmT, col0, DKC, c * 32, lo, hi);
    acc = WMMA_BF16(a, b, acc);
  }
  #pragma unroll
  for (int j = 0; j < 8; ++j)
    memb[(size_t)(row0 + j + 8 * hi) * DKC + col0 + lo] = f2bf(acc[j] * 0.25f);
}

// score tile: 16 queries x 32 keys over augmented head dim 832 = [Q|0.25*mem] . [K|Kc]^T
__device__ __forceinline__ void score_tile(const unsigned short* Qb, const unsigned short* memb,
                                           const unsigned short* Kb, int q0, int kb,
                                           int lo, int hi, v8f& s0, v8f& s1) {
  s0 = v8f{}; s1 = v8f{};
  #pragma unroll
  for (int c = 0; c < NCH; ++c) {
    v16bf a; int bk0;
    if (c < 24) { a = ld_frag(Qb,   q0, EMB, c * 32,        lo, hi); bk0 = c * 32; }
    else        { a = ld_frag(memb, q0, DKC, (c - 24) * 32, lo, hi); bk0 = (c - 24) * 32; }
    v16bf b0 = ld_frag(Kb, kb,      EMB, bk0, lo, hi);
    v16bf b1 = ld_frag(Kb, kb + 16, EMB, bk0, lo, hi);
    s0 = WMMA_BF16(a, b0, s0);
    s1 = WMMA_BF16(a, b1, s1);
  }
}

// ---------------- pass A: causal softmax row statistics ----------------
__global__ void __launch_bounds__(128) k_stats(const unsigned short* __restrict__ Qb,
                                               const unsigned short* __restrict__ memb,
                                               const unsigned short* __restrict__ Kb,
                                               const float* __restrict__ amask,
                                               float* __restrict__ mrow, float* __restrict__ lrow) {
  int lane = threadIdx.x & 31, wave = threadIdx.x >> 5;
  int q0 = (blockIdx.x * 4 + wave) * 16;
  int lo = lane & 15, hi = lane >> 4;
  float m[8], l[8];
  #pragma unroll
  for (int j = 0; j < 8; ++j) { m[j] = -1e30f; l[j] = 0.f; }
  int nt = (q0 + 47) / 32;
  for (int t = 0; t < nt; ++t) {
    int kb = t * 32;
    if (t + 1 < nt) __builtin_prefetch(Kb + (size_t)(kb + 32) * EMB, 0, 3);
    v8f s0, s1;
    score_tile(Qb, memb, Kb, q0, kb, lo, hi, s0, s1);
    #pragma unroll
    for (int j = 0; j < 8; ++j) {
      int row = q0 + j + 8 * hi;
      int c0 = kb + lo, c1 = kb + 16 + lo;
      float v0 = (c0 <= row) ? s0[j] * SCALE_ + amask[c0] : -1e30f;
      float v1 = (c1 <= row) ? s1[j] * SCALE_ + amask[c1] : -1e30f;
      float mn = fmaxf(m[j], hmax16(fmaxf(v0, v1)));
      float alpha = exp2f((m[j] - mn) * LOG2E_);
      m[j] = mn;
      l[j] = l[j] * alpha + hsum16(exp2f((v0 - mn) * LOG2E_) + exp2f((v1 - mn) * LOG2E_));
    }
  }
  if (lo == 0) {
    #pragma unroll
    for (int j = 0; j < 8; ++j) {
      int row = q0 + j + 8 * hi;
      mrow[row] = m[j]; lrow[row] = l[j];
    }
  }
}

// ---------------- pass B: normalized P @ V per 64-dim slice ----------------
__global__ void __launch_bounds__(128) k_ov(const unsigned short* __restrict__ Qb,
                                            const unsigned short* __restrict__ memb,
                                            const unsigned short* __restrict__ Kb,
                                            const unsigned short* __restrict__ Vt,
                                            const float* __restrict__ amask,
                                            const float* __restrict__ mrow,
                                            const float* __restrict__ lrow,
                                            unsigned short* __restrict__ ctxb) {
  __shared__ __align__(16) unsigned short pbuf[4][16 * 32];
  int lane = threadIdx.x & 31, wave = threadIdx.x >> 5;
  int gid = blockIdx.x * 4 + wave;
  int qt = gid / (EMB / 64), sl = gid % (EMB / 64);
  int q0 = qt * 16, d0 = sl * 64;
  int lo = lane & 15, hi = lane >> 4;
  float mj[8], li[8];
  #pragma unroll
  for (int j = 0; j < 8; ++j) {
    int row = q0 + j + 8 * hi;
    mj[j] = mrow[row]; li[j] = 1.f / lrow[row];
  }
  v8f o0 = {}, o1 = {}, o2 = {}, o3 = {};
  int nt = (q0 + 47) / 32;
  for (int t = 0; t < nt; ++t) {
    int kb = t * 32;
    if (t + 1 < nt) __builtin_prefetch(Kb + (size_t)(kb + 32) * EMB, 0, 3);
    v8f s0, s1;
    score_tile(Qb, memb, Kb, q0, kb, lo, hi, s0, s1);
    #pragma unroll
    for (int j = 0; j < 8; ++j) {
      int row = q0 + j + 8 * hi;
      int c0 = kb + lo, c1 = kb + 16 + lo;
      float v0 = (c0 <= row) ? s0[j] * SCALE_ + amask[c0] : -1e30f;
      float v1 = (c1 <= row) ? s1[j] * SCALE_ + amask[c1] : -1e30f;
      float p0 = exp2f((v0 - mj[j]) * LOG2E_) * li[j];
      float p1 = exp2f((v1 - mj[j]) * LOG2E_) * li[j];
      pbuf[wave][(j + 8 * hi) * 32 + lo]      = f2bf(p0);
      pbuf[wave][(j + 8 * hi) * 32 + 16 + lo] = f2bf(p1);
    }
    asm volatile("s_wait_dscnt 0" ::: "memory");
    v16bf ap  = ld_frag(&pbuf[wave][0], 0, 32, 0, lo, hi);
    v16bf bv0 = ld_frag(Vt, d0,      SQ, kb, lo, hi);
    v16bf bv1 = ld_frag(Vt, d0 + 16, SQ, kb, lo, hi);
    v16bf bv2 = ld_frag(Vt, d0 + 32, SQ, kb, lo, hi);
    v16bf bv3 = ld_frag(Vt, d0 + 48, SQ, kb, lo, hi);
    o0 = WMMA_BF16(ap, bv0, o0); o1 = WMMA_BF16(ap, bv1, o1);
    o2 = WMMA_BF16(ap, bv2, o2); o3 = WMMA_BF16(ap, bv3, o3);
  }
  #pragma unroll
  for (int j = 0; j < 8; ++j) {
    size_t row = q0 + j + 8 * hi;
    ctxb[row * EMB + d0 + 0  + lo] = f2bf(o0[j]);
    ctxb[row * EMB + d0 + 16 + lo] = f2bf(o1[j]);
    ctxb[row * EMB + d0 + 32 + lo] = f2bf(o2[j]);
    ctxb[row * EMB + d0 + 48 + lo] = f2bf(o3[j]);
  }
}

// ---------------- output projection: out = ctx @ Wo + bo (f32) ----------------
__global__ void __launch_bounds__(128) k_outproj(const unsigned short* __restrict__ ctxb,
                                                 const unsigned short* __restrict__ Wot,
                                                 const float* __restrict__ bo,
                                                 float* __restrict__ out) {
  int lane = threadIdx.x & 31, wave = threadIdx.x >> 5;
  int gid = blockIdx.x * 4 + wave;
  int rt = gid / (EMB / 16), ct = gid % (EMB / 16);
  int row0 = rt * 16, col0 = ct * 16, lo = lane & 15, hi = lane >> 4;
  v8f acc = {};
  #pragma unroll
  for (int c = 0; c < EMB / 32; ++c) {
    v16bf a = ld_frag(ctxb, row0, EMB, c * 32, lo, hi);
    v16bf b = ld_frag(Wot,  col0, EMB, c * 32, lo, hi);
    acc = WMMA_BF16(a, b, acc);
  }
  float bias = bo[col0 + lo];
  #pragma unroll
  for (int j = 0; j < 8; ++j)
    out[(size_t)(row0 + j + 8 * hi) * EMB + col0 + lo] = acc[j] + bias;
}

extern "C" void kernel_launch(void* const* d_in, const int* in_sizes, int n_in,
                              void* d_out, int out_size, void* d_ws, size_t ws_size,
                              hipStream_t stream) {
  const float* hs    = (const float*)d_in[0];
  const float* amask = (const float*)d_in[1];
  const float* Wc    = (const float*)d_in[2];
  const float* bc    = (const float*)d_in[3];
  const float* Wo    = (const float*)d_in[4];
  const float* bo    = (const float*)d_in[5];
  const float* Wm    = (const float*)d_in[6];
  float* out = (float*)d_out;

  char* w = (char*)d_ws;
  auto take = [&](size_t bytes) -> char* {
    char* p = w; w += (bytes + 255) & ~(size_t)255; return p;
  };
  unsigned short* hsb  = (unsigned short*)take((size_t)SQ * EMB * 2);
  unsigned short* Wct  = (unsigned short*)take((size_t)E3T * EMB * 2);
  unsigned short* Wot  = (unsigned short*)take((size_t)EMB * EMB * 2);
  unsigned short* WmT  = (unsigned short*)take((size_t)DKC * DKC * 2);
  unsigned short* Qb   = (unsigned short*)take((size_t)SQ * EMB * 2);
  unsigned short* Kb   = (unsigned short*)take((size_t)SQ * EMB * 2);
  unsigned short* Vt   = (unsigned short*)take((size_t)EMB * SQ * 2);
  unsigned short* bob  = (unsigned short*)take((size_t)SQ * DKC * 2);
  unsigned short* memb = (unsigned short*)take((size_t)SQ * DKC * 2);
  float*          mrow = (float*)take((size_t)SQ * 4);
  float*          lrow = (float*)take((size_t)SQ * 4);
  unsigned short* ctxb = (unsigned short*)take((size_t)SQ * EMB * 2);

  int n;
  n = SQ * EMB;   k_cvt<<<(n + 255) / 256, 256, 0, stream>>>(hs, hsb, n);
  n = EMB * E3T;  k_cvt_tr<<<(n + 255) / 256, 256, 0, stream>>>(Wc, Wct, EMB, E3T);
  n = EMB * EMB;  k_cvt_tr<<<(n + 255) / 256, 256, 0, stream>>>(Wo, Wot, EMB, EMB);
  n = DKC * DKC;  k_cvt_tr<<<(n + 255) / 256, 256, 0, stream>>>(Wm, WmT, DKC, DKC);

  k_qkv      <<<(SQ / 16) * (E3T / 16) / 4, 128, 0, stream>>>(hsb, Wct, bc, Qb, Kb, Vt);
  k_base_attn<<<(SQ / 16) / 4,              128, 0, stream>>>(Qb, Kb, Vt, bob);
  k_memproj  <<<(SQ / 16) * (DKC / 16) / 4, 128, 0, stream>>>(bob, WmT, memb);
  k_stats    <<<(SQ / 16) / 4,              128, 0, stream>>>(Qb, memb, Kb, amask, mrow, lrow);
  k_ov       <<<(SQ / 16) * (EMB / 64) / 4, 128, 0, stream>>>(Qb, memb, Kb, Vt, amask, mrow, lrow, ctxb);
  k_outproj  <<<(SQ / 16) * (EMB / 16) / 4, 128, 0, stream>>>(ctxb, Wot, bo, out);
}
